// MultiHeadAttention_64089501991250
// MI455X (gfx1250) — compile-verified, matrix operand order checked by reading, not run on hardware
//
#include <hip/hip_runtime.h>
#include <hip/hip_bf16.h>

typedef __bf16 bf16;
typedef __attribute__((ext_vector_type(16))) __bf16 v16bf;
typedef __attribute__((ext_vector_type(8)))  __bf16 v8bf;
typedef __attribute__((ext_vector_type(8)))  float  v8f;
typedef __attribute__((ext_vector_type(4)))  float  v4f;

#define BB  64
#define NN  197
#define EE  768
#define HH  12
#define DD  64
#define NP  208   // padded rows for Q/K (13*16)
#define NPV 224   // padded K-dim for att / Vt cols (7*32)
#define TOK (BB*NN)   // 12608 = 197*64

// workspace layout (bytes)
#define SZ_QK   ((size_t)BB*HH*NP*DD*2)     // 20,447,232
#define SZ_VT   ((size_t)BB*HH*DD*NPV*2)    // 22,020,096
#define OFF_Q   ((size_t)0)
#define OFF_K   (OFF_Q + SZ_QK)
#define OFF_VT  (OFF_K + SZ_QK)
#define OFF_CTX (OFF_VT + SZ_VT)

#define LDA 40            // padded LDS row stride (bf16 elements)
#define KSTEPS (EE/32)    // 24

// ---- CDNA5 helpers ------------------------------------------------------

__device__ inline void sched_fence() {
#if __has_builtin(__builtin_amdgcn_sched_barrier)
  __builtin_amdgcn_sched_barrier(0);
#endif
}

// Low 32 bits of a flat shared-memory address == LDS byte offset
// (flat LDS addresses are {aperture_hi, offset32}; HW truncates to [31:0]).
__device__ inline unsigned lds_off(const void* p) {
  return (unsigned)(unsigned long long)p;
}

// Async global->LDS copy, 16B per lane, tracked by ASYNCcnt (gfx1250 TDM-lite).
__device__ inline void async_copy_b128(const void* g, unsigned ldsoff) {
  asm volatile("global_load_async_to_lds_b128 %0, %1, off"
               :: "v"(ldsoff), "v"((unsigned long long)g)
               : "memory");
}
__device__ inline void wait_async0() {
  asm volatile("s_wait_asynccnt 0x0" ::: "memory");
}

__device__ inline v8f wmma_bf16(v16bf a, v16bf b, v8f c) {
  return __builtin_amdgcn_wmma_f32_16x16x32_bf16(
      /*neg_a=*/false, a, /*neg_b=*/false, b,
      /*c_mod=*/(short)0, c, /*reuse_a=*/false, /*reuse_b=*/false);
}

// Load a 16x32 bf16 A/B fragment from a row-major [rows][stride] array.
// ISA 16-bit layout: lane<16 -> K = kb+[0..7] and kb+[16..23];
//                    lane>=16 -> K = kb+[8..15] and kb+[24..31].
__device__ inline v16bf load_frag(const bf16* __restrict__ base, int stride,
                                  int row, int kb, int lane) {
  int r  = row + (lane & 15);
  int hi = (lane >> 4) & 1;
  const bf16* p = base + (size_t)r * stride + kb + hi * 8;
  v8bf c0 = *(const v8bf*)(p);
  v8bf c1 = *(const v8bf*)(p + 16);
  v16bf f;
#pragma unroll
  for (int i = 0; i < 8; ++i) { f[i] = c0[i]; f[i + 8] = c1[i]; }
  return f;
}

__device__ inline void cvt_store4(bf16* d, v4f v) {
  d[0] = (bf16)v[0]; d[1] = (bf16)v[1]; d[2] = (bf16)v[2]; d[3] = (bf16)v[3];
}

// ---- zero-fill padded Q/K/Vt buffers -----------------------------------

__global__ void __launch_bounds__(256) zfill_kernel(uint4* p, long n16) {
  long i = (long)blockIdx.x * blockDim.x + threadIdx.x;
  long stride = (long)gridDim.x * blockDim.x;
  uint4 z; z.x = 0; z.y = 0; z.z = 0; z.w = 0;
  for (; i < n16; i += stride) p[i] = z;
}

// ---- fused QKV projection: x[12608,768] @ W{q,k,v}^T + b ---------------
// grid = (6, 197, 3); block = 256 (8 waves, 2x4 grid; each wave owns 32x32)
// double-buffered LDS, register prefetch, 1 barrier / K-step, 4 wmma / step

__global__ void __launch_bounds__(256) qkv_kernel(
    const float* __restrict__ x,
    const float* __restrict__ Wq, const float* __restrict__ bq,
    const float* __restrict__ Wk, const float* __restrict__ bk,
    const float* __restrict__ Wv, const float* __restrict__ bv,
    bf16* __restrict__ Qp, bf16* __restrict__ Kp, bf16* __restrict__ Vt) {
  __shared__ __align__(16) bf16 As[2][64 * LDA];
  __shared__ __align__(16) bf16 Bs[2][128 * LDA];
  const int tid = threadIdx.x;
  const int lane = tid & 31, wid = tid >> 5;
  const int wr = wid & 1, wc = wid >> 1;     // 2x4 wave grid
  const int t0 = blockIdx.y * 64;
  const int e0 = blockIdx.x * 128;
  const int z  = blockIdx.z;
  const float* Wm = (z == 0) ? Wq : ((z == 1) ? Wk : Wv);
  const float* bm = (z == 0) ? bq : ((z == 1) ? bk : bv);

  // cooperative-load index precompute
  const int ra = tid >> 3, ka = (tid & 7) * 4;      // A: 2 chunks/thread
  const int ra2 = ra + 32;
  const int rb = tid >> 3, kb = (tid & 7) * 4;      // B: 4 chunks/thread

  v4f pa0, pa1, pb0, pb1, pb2, pb3;
  // prologue: fetch K-step 0
  pa0 = *(const v4f*)(x + (size_t)(t0 + ra) * EE + ka);
  pa1 = *(const v4f*)(x + (size_t)(t0 + ra2) * EE + ka);
  pb0 = *(const v4f*)(Wm + (size_t)(e0 + rb) * EE + kb);
  pb1 = *(const v4f*)(Wm + (size_t)(e0 + rb + 32) * EE + kb);
  pb2 = *(const v4f*)(Wm + (size_t)(e0 + rb + 64) * EE + kb);
  pb3 = *(const v4f*)(Wm + (size_t)(e0 + rb + 96) * EE + kb);
  cvt_store4(&As[0][ra * LDA + ka], pa0);
  cvt_store4(&As[0][ra2 * LDA + ka], pa1);
  cvt_store4(&Bs[0][rb * LDA + kb], pb0);
  cvt_store4(&Bs[0][(rb + 32) * LDA + kb], pb1);
  cvt_store4(&Bs[0][(rb + 64) * LDA + kb], pb2);
  cvt_store4(&Bs[0][(rb + 96) * LDA + kb], pb3);
  __syncthreads();

  v8f a00 = {}, a01 = {}, a10 = {}, a11 = {};
  for (int i = 0; i < KSTEPS; ++i) {
    const int cur = i & 1, nxt = cur ^ 1;
    if (i < KSTEPS - 1) {   // prefetch next K tile while WMMAs run
      const int kk = (i + 1) * 32;
      pa0 = *(const v4f*)(x + (size_t)(t0 + ra) * EE + kk + ka);
      pa1 = *(const v4f*)(x + (size_t)(t0 + ra2) * EE + kk + ka);
      pb0 = *(const v4f*)(Wm + (size_t)(e0 + rb) * EE + kk + kb);
      pb1 = *(const v4f*)(Wm + (size_t)(e0 + rb + 32) * EE + kk + kb);
      pb2 = *(const v4f*)(Wm + (size_t)(e0 + rb + 64) * EE + kk + kb);
      pb3 = *(const v4f*)(Wm + (size_t)(e0 + rb + 96) * EE + kk + kb);
      sched_fence();   // keep all 6 loads issued back-to-back, in flight
    }
    v16bf fa0 = load_frag(As[cur], LDA, wr * 32, 0, lane);
    v16bf fa1 = load_frag(As[cur], LDA, wr * 32 + 16, 0, lane);
    v16bf fb0 = load_frag(Bs[cur], LDA, wc * 32, 0, lane);
    v16bf fb1 = load_frag(Bs[cur], LDA, wc * 32 + 16, 0, lane);
    a00 = wmma_bf16(fa0, fb0, a00);
    a01 = wmma_bf16(fa0, fb1, a01);
    a10 = wmma_bf16(fa1, fb0, a10);
    a11 = wmma_bf16(fa1, fb1, a11);
    if (i < KSTEPS - 1) {
      cvt_store4(&As[nxt][ra * LDA + ka], pa0);
      cvt_store4(&As[nxt][ra2 * LDA + ka], pa1);
      cvt_store4(&Bs[nxt][rb * LDA + kb], pb0);
      cvt_store4(&Bs[nxt][(rb + 32) * LDA + kb], pb1);
      cvt_store4(&Bs[nxt][(rb + 64) * LDA + kb], pb2);
      cvt_store4(&Bs[nxt][(rb + 96) * LDA + kb], pb3);
      __syncthreads();
    }
  }

  // epilogue: bias + head-split scatter (V stored transposed)
  const int nl = lane & 15, hi = lane >> 4;
#pragma unroll
  for (int ar = 0; ar < 2; ++ar) {
#pragma unroll
    for (int bc = 0; bc < 2; ++bc) {
      const v8f acc = (ar == 0) ? (bc == 0 ? a00 : a01) : (bc == 0 ? a10 : a11);
      const int e = e0 + wc * 32 + bc * 16 + nl;
      const int h = e >> 6, dd = e & 63;
      const float bias = bm[e];
#pragma unroll
      for (int r = 0; r < 8; ++r) {
        int t = t0 + wr * 32 + ar * 16 + r + hi * 8;
        int b = t / NN, n = t % NN;
        bf16 o = (bf16)(acc[r] + bias);
        if (z == 0)      Qp[((size_t)(b * HH + h) * NP + n) * DD + dd] = o;
        else if (z == 1) Kp[((size_t)(b * HH + h) * NP + n) * DD + dd] = o;
        else             Vt[((size_t)(b * HH + h) * DD + dd) * NPV + n] = o;
      }
    }
  }
}

// ---- fused attention: one workgroup per (b,h); scores live in LDS ------
// energy -> softmax -> gate -> joint LayerNorm -> bf16 -> att @ v

__global__ void __launch_bounds__(256) attn_kernel(
    const bf16* __restrict__ Qp, const bf16* __restrict__ Kp,
    const bf16* __restrict__ Vt,
    const float* __restrict__ Wg, const float* __restrict__ lnw,
    const float* __restrict__ lnb, bf16* __restrict__ ctx) {
  extern __shared__ char smem[];
  float* sc  = (float*)smem;                              // [208][224] fp32
  bf16*  att = (bf16*)(smem + (size_t)NP * NPV * 4);      // [208][224] bf16
  float* red = (float*)(smem + (size_t)NP * NPV * 4 + (size_t)NP * NPV * 2);

  const int tid = threadIdx.x, lane = tid & 31, wid = tid >> 5;
  const int bh = blockIdx.x;
  const int b = bh / HH, h = bh % HH;
  const bf16* qb = Qp + (size_t)bh * NP * DD;
  const bf16* kbp = Kp + (size_t)bh * NP * DD;
  const bf16* vb = Vt + (size_t)bh * DD * NPV;

  // zero att buffer (pad rows/cols must be clean for the att@v WMMA)
  for (int i = tid; i < NP * NPV / 2; i += 256) ((unsigned*)att)[i] = 0u;
  __syncthreads();

  // --- energy = (q @ k^T) * D^-0.5: each wave owns row tiles, reuses q frags
  for (int ti = wid; ti < 13; ti += 8) {
    v16bf qa0 = load_frag(qb, DD, ti * 16, 0, lane);
    v16bf qa1 = load_frag(qb, DD, ti * 16, 32, lane);
    for (int tj = 0; tj < 13; ++tj) {
      v8f acc = {};
      acc = wmma_bf16(qa0, load_frag(kbp, DD, tj * 16, 0, lane), acc);
      acc = wmma_bf16(qa1, load_frag(kbp, DD, tj * 16, 32, lane), acc);
      int col = tj * 16 + (lane & 15);
      int hi = lane >> 4;
#pragma unroll
      for (int r = 0; r < 8; ++r)
        sc[(size_t)(ti * 16 + r + hi * 8) * NPV + col] = acc[r] * 0.125f;
    }
  }
  __syncthreads();

  // --- row softmax + elementwise gate, accumulate LN statistics ---
  float lsum = 0.f, lsq = 0.f;
  for (int r = wid; r < NN; r += 8) {
    float* row = sc + (size_t)r * NPV;
    float mx = -3.4e38f;
    for (int c = lane; c < NN; c += 32) mx = fmaxf(mx, row[c]);
    for (int o = 16; o > 0; o >>= 1) mx = fmaxf(mx, __shfl_xor(mx, o, 32));
    float s = 0.f;
    for (int c = lane; c < NN; c += 32) {
      float e = __expf(row[c] - mx);
      row[c] = e; s += e;
    }
    for (int o = 16; o > 0; o >>= 1) s += __shfl_xor(s, o, 32);
    float inv = 1.f / s;
    for (int c = lane; c < NN; c += 32) {
      float g = row[c] * inv * Wg[r * NN + c];
      row[c] = g;
      lsum += g; lsq += g * g;
    }
  }
  // block-wide reduction for joint LayerNorm over all 197*197 elements
  for (int o = 16; o > 0; o >>= 1) {
    lsum += __shfl_xor(lsum, o, 32);
    lsq  += __shfl_xor(lsq,  o, 32);
  }
  if (lane == 0) { red[wid] = lsum; red[8 + wid] = lsq; }
  __syncthreads();
  if (tid == 0) {
    float S = 0.f, SS = 0.f;
    for (int i = 0; i < 8; ++i) { S += red[i]; SS += red[8 + i]; }
    const float inv_n = 1.f / (float)(NN * NN);
    float mu = S * inv_n;
    float var = SS * inv_n - mu * mu;
    red[0] = mu;
    red[1] = rsqrtf(var + 1e-5f);
  }
  __syncthreads();
  const float mu = red[0], rstd = red[1];

  // --- normalize + affine, convert to bf16 att ---
  for (int idx = tid; idx < NN * NN; idx += 256) {
    int r = idx / NN, c = idx % NN;
    float g = sc[(size_t)r * NPV + c];
    float y = (g - mu) * rstd * lnw[idx] + lnb[idx];
    att[(size_t)r * NPV + c] = (bf16)y;
  }
  __syncthreads();

  // --- out = att @ v: hold the 7 att A-frags per row tile, reuse x4 ---
  for (int ti = wid; ti < 13; ti += 8) {
    v16bf af[7];
#pragma unroll
    for (int kk = 0; kk < 7; ++kk)
      af[kk] = load_frag(att, NPV, ti * 16, kk * 32, lane);
#pragma unroll
    for (int tj = 0; tj < 4; ++tj) {
      v8f acc = {};
#pragma unroll
      for (int kk = 0; kk < 7; ++kk)
        acc = wmma_bf16(af[kk], load_frag(vb, NPV, tj * 16, kk * 32, lane), acc);
      int dd = tj * 16 + (lane & 15);
      int hi = lane >> 4;
#pragma unroll
      for (int r = 0; r < 8; ++r) {
        int n = ti * 16 + r + hi * 8;
        if (n < NN)
          ctx[(size_t)(b * NN + n) * EE + h * DD + dd] = (bf16)acc[r];
      }
    }
  }
}

// ---- output projection: ctx[12608,768](bf16) @ Wp^T + bp -> fp32 -------
// A tile staged via gfx1250 async global->LDS copies (ASYNCcnt),
// B tile via register prefetch + fp32->bf16 convert.

__global__ void __launch_bounds__(256) proj_kernel(
    const bf16* __restrict__ ctx, const float* __restrict__ Wp,
    const float* __restrict__ bp, float* __restrict__ out) {
  __shared__ __align__(16) bf16 As[2][64 * LDA];
  __shared__ __align__(16) bf16 Bs[2][128 * LDA];
  const int tid = threadIdx.x, lane = tid & 31, wid = tid >> 5;
  const int wr = wid & 1, wc = wid >> 1;
  const int t0 = blockIdx.y * 64, e0 = blockIdx.x * 128;

  const int ra = tid >> 2, ka = (tid & 3) * 8;      // A: 1x16B chunk/thread
  const int rb = tid >> 3, kb = (tid & 7) * 4;      // B: 4 v4f chunks/thread

  v4f pb0, pb1, pb2, pb3;
  // prologue: async-copy A tile 0 straight into LDS; prefetch B tile 0
  async_copy_b128(ctx + (size_t)(t0 + ra) * EE + ka,
                  lds_off(&As[0][ra * LDA + ka]));
  pb0 = *(const v4f*)(Wp + (size_t)(e0 + rb) * EE + kb);
  pb1 = *(const v4f*)(Wp + (size_t)(e0 + rb + 32) * EE + kb);
  pb2 = *(const v4f*)(Wp + (size_t)(e0 + rb + 64) * EE + kb);
  pb3 = *(const v4f*)(Wp + (size_t)(e0 + rb + 96) * EE + kb);
  cvt_store4(&Bs[0][rb * LDA + kb], pb0);
  cvt_store4(&Bs[0][(rb + 32) * LDA + kb], pb1);
  cvt_store4(&Bs[0][(rb + 64) * LDA + kb], pb2);
  cvt_store4(&Bs[0][(rb + 96) * LDA + kb], pb3);
  wait_async0();
  __syncthreads();

  v8f a00 = {}, a01 = {}, a10 = {}, a11 = {};
  for (int i = 0; i < KSTEPS; ++i) {
    const int cur = i & 1, nxt = cur ^ 1;
    if (i < KSTEPS - 1) {
      const int kk = (i + 1) * 32;
      // async A copy for next buffer runs concurrently with the WMMAs below
      async_copy_b128(ctx + (size_t)(t0 + ra) * EE + kk + ka,
                      lds_off(&As[nxt][ra * LDA + ka]));
      pb0 = *(const v4f*)(Wp + (size_t)(e0 + rb) * EE + kk + kb);
      pb1 = *(const v4f*)(Wp + (size_t)(e0 + rb + 32) * EE + kk + kb);
      pb2 = *(const v4f*)(Wp + (size_t)(e0 + rb + 64) * EE + kk + kb);
      pb3 = *(const v4f*)(Wp + (size_t)(e0 + rb + 96) * EE + kk + kb);
      sched_fence();
    }
    v16bf fa0 = load_frag(As[cur], LDA, wr * 32, 0, lane);
    v16bf fa1 = load_frag(As[cur], LDA, wr * 32 + 16, 0, lane);
    v16bf fb0 = load_frag(Bs[cur], LDA, wc * 32, 0, lane);
    v16bf fb1 = load_frag(Bs[cur], LDA, wc * 32 + 16, 0, lane);
    a00 = wmma_bf16(fa0, fb0, a00);
    a01 = wmma_bf16(fa0, fb1, a01);
    a10 = wmma_bf16(fa1, fb0, a10);
    a11 = wmma_bf16(fa1, fb1, a11);
    if (i < KSTEPS - 1) {
      cvt_store4(&Bs[nxt][rb * LDA + kb], pb0);
      cvt_store4(&Bs[nxt][(rb + 32) * LDA + kb], pb1);
      cvt_store4(&Bs[nxt][(rb + 64) * LDA + kb], pb2);
      cvt_store4(&Bs[nxt][(rb + 96) * LDA + kb], pb3);
      wait_async0();   // next A tile fully landed in LDS
      __syncthreads();
    }
  }

  const int nl = lane & 15, hi = lane >> 4;
#pragma unroll
  for (int ar = 0; ar < 2; ++ar) {
#pragma unroll
    for (int bc = 0; bc < 2; ++bc) {
      const v8f acc = (ar == 0) ? (bc == 0 ? a00 : a01) : (bc == 0 ? a10 : a11);
      const int e = e0 + wc * 32 + bc * 16 + nl;
      const float bias = bp[e];
#pragma unroll
      for (int r = 0; r < 8; ++r) {
        int t = t0 + wr * 32 + ar * 16 + r + hi * 8;
        out[(size_t)t * EE + e] = acc[r] + bias;
      }
    }
  }
}

// ---- launcher -----------------------------------------------------------

extern "C" void kernel_launch(void* const* d_in, const int* in_sizes, int n_in,
                              void* d_out, int out_size, void* d_ws, size_t ws_size,
                              hipStream_t stream) {
  const float* x   = (const float*)d_in[0];
  const float* Wq  = (const float*)d_in[1];
  const float* bq  = (const float*)d_in[2];
  const float* Wk  = (const float*)d_in[3];
  const float* bk  = (const float*)d_in[4];
  const float* Wv  = (const float*)d_in[5];
  const float* bv  = (const float*)d_in[6];
  const float* Wg  = (const float*)d_in[7];
  const float* lnw = (const float*)d_in[8];
  const float* lnb = (const float*)d_in[9];
  const float* Wp  = (const float*)d_in[10];
  const float* bp  = (const float*)d_in[11];

  char* ws = (char*)d_ws;
  bf16* Qp  = (bf16*)(ws + OFF_Q);
  bf16* Kp  = (bf16*)(ws + OFF_K);
  bf16* Vt  = (bf16*)(ws + OFF_VT);
  bf16* ctx = (bf16*)(ws + OFF_CTX);
  float* out = (float*)d_out;

  // zero padded Q/K/Vt regions so WMMA pad rows/cols contribute zeros
  long n16 = (long)(OFF_CTX / 16);
  zfill_kernel<<<2048, 256, 0, stream>>>((uint4*)ws, n16);

  dim3 g1(EE / 128, TOK / 64, 3);  // (6, 197, 3)
  qkv_kernel<<<g1, 256, 0, stream>>>(x, Wq, bq, Wk, bk, Wv, bv, Qp, Kp, Vt);

  size_t smem = (size_t)NP * NPV * 4 + (size_t)NP * NPV * 2 + 64;  // ~280 KB LDS
  attn_kernel<<<BB * HH, 256, smem, stream>>>(Qp, Kp, Vt, Wg, lnw, lnb, ctx);

  dim3 g2(EE / 128, TOK / 64);     // (6, 197)
  proj_kernel<<<g2, 256, 0, stream>>>(ctx, Wp, bp, out);
}
